// _Interactions_3856880632376
// MI455X (gfx1250) — matrix operand order repaired and checked by reading, not compiled, not run on hardware
//
#include <hip/hip_runtime.h>
#include <hip/hip_bf16.h>

// ---------------------------------------------------------------------------
// Types for CDNA5 WMMA
// ---------------------------------------------------------------------------
typedef __bf16 bf16_t;
typedef bf16_t v16bf __attribute__((ext_vector_type(16)));
typedef bf16_t v8bf  __attribute__((ext_vector_type(8)));
typedef float  v8f   __attribute__((ext_vector_type(8)));

#define HIDC 128        // HID == C == G == 128
#define AT_STRIDE 136   // staged A row stride in bf16 elems (128 + 8 pad -> bank spread)

#define LOG2E 1.4426950408889634f
#define LN2   0.6931471805599453f

// Hardware transcendentals (v_exp_f32 = 2^x, v_log_f32 = log2, v_rcp_f32).
__device__ __forceinline__ float fast_exp(float x) {
    return __builtin_amdgcn_exp2f(x * LOG2E);
}
// softplus(x) = max(x,0) + ln(1 + e^{-|x|}); argument of log in [1,2] -> hw log is accurate
__device__ __forceinline__ float softplus_f(float x) {
    float t = __builtin_amdgcn_exp2f(-fabsf(x) * LOG2E);
    return fmaxf(x, 0.0f) + __builtin_amdgcn_logf(1.0f + t) * LN2;
}
// sigmoid(x) = 1/(1+e^{-x}); denominator in (1, +inf), rcp safe
__device__ __forceinline__ float sigmoid_f(float x) {
    return __builtin_amdgcn_rcpf(1.0f + fast_exp(-x));
}

// ---------------------------------------------------------------------------
// Pack fp32 weights into per-lane bf16 WMMA B-fragments.
// B frag k order used here (contiguous, same as A): lane<16: N=lane,   K=k0+i (i=0..15)
//                                  lane>=16:N=lane-16,K=k0+16+i
// Dest layout: [colblock][kstep(4)][lane(32)][i(16)]  (32B contiguous per lane)
// colblock-groups: 0..7 -> W0 ; 8..15 -> Wf rows[0:128] ; 16..23 -> Wf rows[128:256]
//                  24..31 -> Ws rows[0:128] ; 32..39 -> Ws rows[128:256]
// ---------------------------------------------------------------------------
__global__ void pack_weights_kernel(const float* __restrict__ W0,
                                    const float* __restrict__ Wf,
                                    const float* __restrict__ Ws,
                                    bf16_t* __restrict__ w0pack,
                                    bf16_t* __restrict__ wcatpack) {
    int gid  = blockIdx.x * 256 + threadIdx.x;      // 0 .. 81919
    int i    = gid & 15;
    int lane = (gid >> 4) & 31;
    int ks   = (gid >> 9) & 3;
    int cbg  = gid >> 11;                           // 0..39
    int K    = ks * 32 + ((lane >= 16) ? 16 : 0) + i;
    int n    = lane & 15;
    float v;
    bf16_t* dst;
    if (cbg < 8) {
        v   = W0[K * HIDC + cbg * 16 + n];
        dst = w0pack + gid;
    } else {
        int g  = (cbg - 8) >> 3;                    // 0..3
        int cb = (cbg - 8) & 7;
        const float* Wsrc = (g < 2) ? Wf : Ws;
        int krow = K + ((g & 1) ? HIDC : 0);
        v   = Wsrc[krow * HIDC + cb * 16 + n];
        dst = wcatpack + (gid - 8 * 2048);
    }
    *dst = (bf16_t)v;
}

// ---------------------------------------------------------------------------
// Zero scratch (aggr + stat sums)
// ---------------------------------------------------------------------------
__global__ void zero_kernel(float* __restrict__ p, int n) {
    int gid = blockIdx.x * 256 + threadIdx.x;
    if (gid < n) p[gid] = 0.0f;
}

// ---------------------------------------------------------------------------
// GEMM: C = act( A[M,128] @ Bpack ), bf16 WMMA, fp32 accumulate.
// Block = 256 threads (8 waves) handles a 32-row tile:
//   - stage A tile fp32->bf16 into LDS once (coop, padded stride)
//   - each wave owns colblock cb = blockIdx.y*8 + wave and computes TWO 16x16
//     subtiles (rows [0:16) and [16:32) of the tile), reusing one B fragment.
// mode 0: add bias + softplus (lin0).  mode 1: raw (projection table).
// A-frag (contiguous k order used here, same as Bpack): M=lane&15, klane=(lane>=16)?16:0,
//                          K = k0 + klane + i
// ---------------------------------------------------------------------------
__global__ void gemm_bf16_wmma_kernel(const float* __restrict__ A,
                                      const bf16_t* __restrict__ Bpack,
                                      const float* __restrict__ bias,
                                      float* __restrict__ C,
                                      int ldc, int mode, int nrowsTotal) {
    __shared__ __align__(32) bf16_t atile[32 * AT_STRIDE];

    const int tid   = threadIdx.x;
    const int lane  = tid & 31;
    const int wave  = tid >> 5;
    const int row0  = blockIdx.x * 32;
    const int cb    = blockIdx.y * 8 + wave;
    const int rem   = nrowsTotal - row0;            // rows in this tile (may be 16)

    // ---- cooperative stage: 32 rows x 128 cols, fp32 -> bf16 ----
    {
        int r = tid >> 3;                // 0..31
        int p = (tid & 7) * 16;          // 0,16,...,112
        v8bf h0, h1;
        if (r < rem) {
            const float* src = A + (size_t)(row0 + r) * HIDC + p;
            float4 f0 = *(const float4*)(src + 0);
            float4 f1 = *(const float4*)(src + 4);
            float4 f2 = *(const float4*)(src + 8);
            float4 f3 = *(const float4*)(src + 12);
            h0[0] = (bf16_t)f0.x; h0[1] = (bf16_t)f0.y; h0[2] = (bf16_t)f0.z; h0[3] = (bf16_t)f0.w;
            h0[4] = (bf16_t)f1.x; h0[5] = (bf16_t)f1.y; h0[6] = (bf16_t)f1.z; h0[7] = (bf16_t)f1.w;
            h1[0] = (bf16_t)f2.x; h1[1] = (bf16_t)f2.y; h1[2] = (bf16_t)f2.z; h1[3] = (bf16_t)f2.w;
            h1[4] = (bf16_t)f3.x; h1[5] = (bf16_t)f3.y; h1[6] = (bf16_t)f3.z; h1[7] = (bf16_t)f3.w;
        } else {
            h0 = (v8bf)(bf16_t)0.0f;
            h1 = (v8bf)(bf16_t)0.0f;
        }
        *(v8bf*)(atile + r * AT_STRIDE + p)     = h0;
        *(v8bf*)(atile + r * AT_STRIDE + p + 8) = h1;
    }
    __syncthreads();

    const int klane = (lane >= 16) ? 16 : 0;
    const int mrow  = lane & 15;

    v8f acc0 = {};
    v8f acc1 = {};
#pragma unroll
    for (int ks = 0; ks < 4; ++ks) {
        const int koff = ks * 32 + klane;
        // B fragment: one contiguous 32B load, shared by both subtiles
        v16bf b = *(const v16bf*)(Bpack + (((size_t)cb * 4 + ks) * 32 + lane) * 16);
        // subtile 0 fragment (rows 0..15 of tile)
        v8bf lo0 = *(const v8bf*)(atile + mrow * AT_STRIDE + koff);
        v8bf hi0 = *(const v8bf*)(atile + mrow * AT_STRIDE + koff + 8);
        v16bf a0 = __builtin_shufflevector(lo0, hi0, 0,1,2,3,4,5,6,7,8,9,10,11,12,13,14,15);
        // subtile 1 fragment (rows 16..31 of tile)
        v8bf lo1 = *(const v8bf*)(atile + (mrow + 16) * AT_STRIDE + koff);
        v8bf hi1 = *(const v8bf*)(atile + (mrow + 16) * AT_STRIDE + koff + 8);
        v16bf a1 = __builtin_shufflevector(lo1, hi1, 0,1,2,3,4,5,6,7,8,9,10,11,12,13,14,15);

        acc0 = __builtin_amdgcn_wmma_f32_16x16x32_bf16(false, a0, false, b,
                                                       (short)0, acc0, false, false);
        acc1 = __builtin_amdgcn_wmma_f32_16x16x32_bf16(false, a1, false, b,
                                                       (short)0, acc1, false, false);
    }

    // C/D layout: lane<16: N=lane, M=r ; lane>=16: N=lane-16, M=8+r
    const int n     = lane & 15;
    const int mbase = (lane >= 16) ? 8 : 0;
    const int col   = cb * 16 + n;
    const float bv  = (mode == 0) ? bias[col] : 0.0f;
#pragma unroll
    for (int r = 0; r < 8; ++r) {
        int row = row0 + mbase + r;
        if (row < nrowsTotal) {
            float v = acc0[r] + bv;
            if (mode == 0) v = softplus_f(v);
            C[(size_t)row * ldc + col] = v;
        }
        int row1 = row + 16;
        if (row1 < nrowsTotal) {
            float v = acc1[r] + bv;
            if (mode == 0) v = softplus_f(v);
            C[(size_t)row1 * ldc + col] = v;
        }
    }
}

// ---------------------------------------------------------------------------
// Edge kernel: one wave per edge iteration (4 channels per lane), grid-stride
// with software prefetch of the next edge's node rows (L2-resident gathers).
//   e    = softplus(edge_attr[e] . Wsh + bsh)        (wave reduction)
//   gate = sigmoid (pi_f[dst] + pj_f[src] + e*WfE + bf)
//   core = softplus(pi_s[dst] + pj_s[src] + e*WsE + bs)
//   aggr[dst] += gate*core                           (f32 atomics)
// P layout: [N,512] = [pi_f | pj_f | pi_s | pj_s]
// ---------------------------------------------------------------------------
__global__ void edge_kernel(const int* __restrict__ ei,
                            const float* __restrict__ edge_attr,
                            const float* __restrict__ Wsh,
                            const float* __restrict__ bsh,
                            const float* __restrict__ Wf,
                            const float* __restrict__ bf,
                            const float* __restrict__ Ws,
                            const float* __restrict__ bs,
                            const float* __restrict__ P,
                            float* __restrict__ aggr,
                            int E, int waveCount) {
    const int lane = threadIdx.x & 31;
    const int w    = blockIdx.x * 8 + (threadIdx.x >> 5);
    const int c0   = lane * 4;

    // loop-invariant rows (L1-resident)
    float4 ww  = *(const float4*)(Wsh + c0);
    float4 wfe = *(const float4*)(Wf + 256 * HIDC + c0);
    float4 wse = *(const float4*)(Ws + 256 * HIDC + c0);
    float4 bfv = *(const float4*)(bf + c0);
    float4 bsv = *(const float4*)(bs + c0);
    const float bshv = bsh[0];

    for (int e = w; e < E; e += waveCount) {
        const int s = ei[e];        // source j
        const int d = ei[E + e];    // target i (aggregation index)

        // prefetch next iteration's gather rows into cache
        int en = e + waveCount;
        if (en < E) {
            int sn = ei[en];
            int dn = ei[E + en];
            __builtin_prefetch(P + (size_t)dn * 512 + c0, 0, 0);
            __builtin_prefetch(P + (size_t)sn * 512 + 128 + c0, 0, 0);
            __builtin_prefetch(P + (size_t)dn * 512 + 256 + c0, 0, 0);
            __builtin_prefetch(P + (size_t)sn * 512 + 384 + c0, 0, 0);
        }

        // 128-element dot product across the wave
        float4 ea = *(const float4*)(edge_attr + (size_t)e * HIDC + c0);
        float dotv = ea.x * ww.x + ea.y * ww.y + ea.z * ww.z + ea.w * ww.w;
#pragma unroll
        for (int off = 16; off >= 1; off >>= 1)
            dotv += __shfl_xor(dotv, off, 32);
        const float ev = softplus_f(dotv + bshv);

        const float* Pi = P + (size_t)d * 512;
        const float* Pj = P + (size_t)s * 512;
        float4 pif = *(const float4*)(Pi + c0);
        float4 pjf = *(const float4*)(Pj + 128 + c0);
        float4 pis = *(const float4*)(Pi + 256 + c0);
        float4 pjs = *(const float4*)(Pj + 384 + c0);

        float g0 = sigmoid_f(pif.x + pjf.x + ev * wfe.x + bfv.x);
        float g1 = sigmoid_f(pif.y + pjf.y + ev * wfe.y + bfv.y);
        float g2 = sigmoid_f(pif.z + pjf.z + ev * wfe.z + bfv.z);
        float g3 = sigmoid_f(pif.w + pjf.w + ev * wfe.w + bfv.w);
        float h0 = softplus_f(pis.x + pjs.x + ev * wse.x + bsv.x);
        float h1 = softplus_f(pis.y + pjs.y + ev * wse.y + bsv.y);
        float h2 = softplus_f(pis.z + pjs.z + ev * wse.z + bsv.z);
        float h3 = softplus_f(pis.w + pjs.w + ev * wse.w + bsv.w);

        float* ag = aggr + (size_t)d * HIDC + c0;
        atomicAdd(ag + 0, g0 * h0);
        atomicAdd(ag + 1, g1 * h1);
        atomicAdd(ag + 2, g2 * h2);
        atomicAdd(ag + 3, g3 * h3);
    }
}

// ---------------------------------------------------------------------------
// BatchNorm statistics: per-channel sum / sumsq (block-local accumulate, then
// one atomic per block per channel). blockDim = 128 (channel = threadIdx.x).
// ---------------------------------------------------------------------------
__global__ void bn_stats_kernel(const float* __restrict__ aggr,
                                float* __restrict__ sums, int N) {
    const int c = threadIdx.x;
    float s = 0.0f, sq = 0.0f;
    for (int n = blockIdx.x; n < N; n += gridDim.x) {
        float v = aggr[(size_t)n * HIDC + c];
        s  += v;
        sq += v * v;
    }
    atomicAdd(&sums[c], s);
    atomicAdd(&sums[HIDC + c], sq);
}

// ---------------------------------------------------------------------------
// Final epilogue: result = 2*out + gamma*(aggr-mean)*rsqrt(var+eps) + beta
// ---------------------------------------------------------------------------
__global__ void finalize_kernel(const float* __restrict__ out,
                                const float* __restrict__ aggr,
                                const float* __restrict__ sums,
                                const float* __restrict__ gamma,
                                const float* __restrict__ beta,
                                float* __restrict__ res,
                                int total, float invN) {
    int gid = blockIdx.x * 256 + threadIdx.x;
    if (gid >= total) return;
    int c = gid & (HIDC - 1);
    float mean = sums[c] * invN;
    float var  = sums[HIDC + c] * invN - mean * mean;
    float inv  = rsqrtf(var + 1e-5f);
    res[gid] = 2.0f * out[gid] + (aggr[gid] - mean) * inv * gamma[c] + beta[c];
}

// ---------------------------------------------------------------------------
// Launcher
// ---------------------------------------------------------------------------
extern "C" void kernel_launch(void* const* d_in, const int* in_sizes, int n_in,
                              void* d_out, int out_size, void* d_ws, size_t ws_size,
                              hipStream_t stream) {
    const float* h         = (const float*)d_in[0];
    const int*   ei        = (const int*)d_in[1];
    /* d_in[2] = edge_weight : unused by the reference */
    const float* edge_attr = (const float*)d_in[3];
    const float* W0        = (const float*)d_in[4];
    const float* b0        = (const float*)d_in[5];
    const float* Wsh       = (const float*)d_in[6];
    const float* bsh       = (const float*)d_in[7];
    const float* Wf        = (const float*)d_in[8];
    const float* bf        = (const float*)d_in[9];
    const float* Ws        = (const float*)d_in[10];
    const float* bs        = (const float*)d_in[11];
    const float* gamma     = (const float*)d_in[12];
    const float* beta      = (const float*)d_in[13];

    const int N  = in_sizes[0] / HIDC;        // 50000
    const int E  = in_sizes[1] / 2;           // 600000
    const int NC = N * HIDC;                  // 6,400,000

    // Workspace layout (floats)
    float* ws    = (float*)d_ws;
    float* out   = ws;                        // [N,128]
    float* P     = ws + (size_t)NC;           // [N,512]
    float* aggr  = ws + (size_t)NC * 5;       // [N,128]
    float* sums  = ws + (size_t)NC * 6;       // [256]
    bf16_t* w0pack   = (bf16_t*)(ws + (size_t)NC * 6 + 256);
    bf16_t* wcatpack = w0pack + 8 * 4 * 32 * 16;   // +16384 bf16

    // 1) pack weights into WMMA B-fragment order (81920 elems)
    pack_weights_kernel<<<320, 256, 0, stream>>>(W0, Wf, Ws, w0pack, wcatpack);

    // 2) zero aggr + stat sums
    zero_kernel<<<(NC + 256 + 255) / 256, 256, 0, stream>>>(aggr, NC + 256);

    // 3) out = softplus(h @ W0 + b0)   -- bf16 WMMA, LDS-staged A
    const int rt32 = (N + 31) / 32;
    dim3 g1(rt32, 1);
    gemm_bf16_wmma_kernel<<<g1, 256, 0, stream>>>(h, w0pack, b0, out, HIDC, 0, N);

    // 4) P = out @ [Wf_i | Wf_j | Ws_i | Ws_j]  -> [N,512]  -- bf16 WMMA
    dim3 g2(rt32, 4);
    gemm_bf16_wmma_kernel<<<g2, 256, 0, stream>>>(out, wcatpack, nullptr, P, 512, 1, N);

    // 5) edge gather / gated message / scatter-add (L2-resident tables),
    //    grid-stride with next-edge prefetch
    const int edgeBlocks = 18750;                 // 150000 waves, 4 edges/wave
    edge_kernel<<<edgeBlocks, 256, 0, stream>>>(ei, edge_attr, Wsh, bsh,
                                                Wf, bf, Ws, bs, P, aggr,
                                                E, edgeBlocks * 8);

    // 6) batch-norm statistics
    bn_stats_kernel<<<512, HIDC, 0, stream>>>(aggr, sums, N);

    // 7) result = 2*out + BN(aggr)
    finalize_kernel<<<(NC + 255) / 256, 256, 0, stream>>>(
        out, aggr, sums, gamma, beta, (float*)d_out, NC, 1.0f / (float)N);
}
